// MultiHeadAttention_50379966382939
// MI455X (gfx1250) — compile-verified, matrix operand order checked by reading, not run on hardware
//
#include <hip/hip_runtime.h>
#include <hip/hip_bf16.h>

// ---------------------------------------------------------------------------
// MHA on gfx1250: bf16 WMMA (v_wmma_f32_16x16x32_bf16) for all GEMMs,
// flash-attention with online softmax, f32 accumulation, block-cooperative
// async K/V staging into LDS (GLOBAL_LOAD_ASYNC_TO_LDS_B128 when available).
// ---------------------------------------------------------------------------

typedef __attribute__((ext_vector_type(16))) __bf16 v16bf;
typedef __attribute__((ext_vector_type(8)))  __bf16 v8bf;
typedef __attribute__((ext_vector_type(4)))  __bf16 v4bf;
typedef __attribute__((ext_vector_type(8)))  float  v8f;
typedef __attribute__((ext_vector_type(4)))  float  v4f;
typedef __attribute__((ext_vector_type(4)))  int    v4i;

#define B_  2
#define T_  2048
#define D_  1024
#define H_  16
#define DK_ 64

#if defined(__has_builtin)
#if __has_builtin(__builtin_amdgcn_global_load_async_to_lds_b128)
#define HAS_ASYNC_LDS 1
#endif
#if __has_builtin(__builtin_amdgcn_s_wait_asynccnt)
#define HAS_ASYNC_WAIT 1
#endif
#endif

typedef __attribute__((address_space(1))) v4i* gptr_v4i;
typedef __attribute__((address_space(3))) v4i* lptr_v4i;

__device__ __forceinline__ v8f zero8() {
  v8f z;
#pragma unroll
  for (int i = 0; i < 8; ++i) z[i] = 0.0f;
  return z;
}

// 16-byte async copy global -> LDS (falls back to load+store if builtin absent)
__device__ __forceinline__ void async_copy16(const __bf16* g, __bf16* l) {
#if defined(HAS_ASYNC_LDS)
  __builtin_amdgcn_global_load_async_to_lds_b128((gptr_v4i)g, (lptr_v4i)l, 0,
                                                 0);
#else
  *(v8bf*)l = *(const v8bf*)g;
#endif
}

__device__ __forceinline__ void async_wait_all() {
#if defined(HAS_ASYNC_LDS)
#if defined(HAS_ASYNC_WAIT)
  __builtin_amdgcn_s_wait_asynccnt(0);
#else
  asm volatile("s_wait_asynccnt 0x0" ::: "memory");
#endif
#endif
}

// Load a 16-bit WMMA fragment (A 16x32, or B 32x16 given transposed storage).
// CDNA5 16-bit fragment layout: lane = (r | g<<4), r = row(A)/col(B),
// elements cover K in [8g,8g+8) and [16+8g,16+8g+8) -> two b128 loads.
__device__ __forceinline__ v16bf load_frag(const __bf16* base, int ld) {
  const int lane = threadIdx.x & 31;
  const int r = lane & 15;
  const int g = lane >> 4;
  const __bf16* p = base + r * ld + 8 * g;
  v8bf c0 = *(const v8bf*)(p);
  v8bf c1 = *(const v8bf*)(p + 16);
  v16bf a;
#pragma unroll
  for (int e = 0; e < 8; ++e) { a[e] = c0[e]; a[e + 8] = c1[e]; }
  return a;
}

__device__ __forceinline__ v8f wmma_bf16(v16bf a, v16bf b, v8f c) {
  return __builtin_amdgcn_wmma_f32_16x16x32_bf16(false, a, false, b,
                                                 (short)0, c, false, false);
}

// ---------------------------------------------------------------------------
// fp32 -> bf16 elementwise convert (vectorized x4)
// ---------------------------------------------------------------------------
__global__ __launch_bounds__(256) void cvt_bf16_kernel(
    const float* __restrict__ in, __bf16* __restrict__ out, int n) {
  int i = (blockIdx.x * 256 + threadIdx.x) * 4;
  if (i < n) {
    v4f f = *(const v4f*)(in + i);
    v4bf o;
#pragma unroll
    for (int e = 0; e < 4; ++e) o[e] = (__bf16)f[e];
    *(v4bf*)(out + i) = o;
  }
}

// ---------------------------------------------------------------------------
// fp32 DxD weight -> transposed bf16 (Wt[n][k] = W[k][n]) via LDS tile
// ---------------------------------------------------------------------------
__global__ __launch_bounds__(256) void wtrans_kernel(
    const float* __restrict__ in, __bf16* __restrict__ out) {
  __shared__ float tile[32][33];
  const int x = blockIdx.x * 32 + threadIdx.x;   // column n of W
  const int y0 = blockIdx.y * 32;                // row k of W
#pragma unroll
  for (int r = threadIdx.y; r < 32; r += 8)
    tile[r][threadIdx.x] = in[(y0 + r) * D_ + x];
  __syncthreads();
  const int xo = y0 + threadIdx.x;               // k index in output row
#pragma unroll
  for (int r = threadIdx.y; r < 32; r += 8)
    out[(blockIdx.x * 32 + r) * D_ + xo] = (__bf16)tile[threadIdx.x][r];
}

// ---------------------------------------------------------------------------
// bf16 GEMM: C[m][n] = sum_k A[m][k] * Bt[n][k]     (Bt = B transposed)
// Block: 256 threads = 8 waves, 64(M) x 128(N) block tile, 16x64 per wave.
// MODE 0: out bf16, head-split row-major  [(b,h,t),dk]
// MODE 1: out bf16, head-split transposed [(b,h,dk),t]
// MODE 2: out f32 row-major [m][n]
// ---------------------------------------------------------------------------
template <int MODE>
__global__ __launch_bounds__(256) void gemm_bf16_kernel(
    const __bf16* __restrict__ A, const __bf16* __restrict__ Bt,
    void* __restrict__ out, int M, int N, int Kd) {
  const int w    = threadIdx.x >> 5;
  const int lane = threadIdx.x & 31;
  const int c    = lane & 15;
  const int g    = lane >> 4;
  const int mw   = w & 3;
  const int nw   = w >> 2;
  const int row0 = blockIdx.y * 64 + mw * 16;
  const int n0   = blockIdx.x * 128 + nw * 64;

  v8f acc[4];
#pragma unroll
  for (int t = 0; t < 4; ++t) acc[t] = zero8();

  const __bf16* pa = A + (size_t)row0 * Kd;
  for (int k0 = 0; k0 < Kd; k0 += 32) {
    __builtin_prefetch(pa + k0 + 128, 0, 1);           // global_prefetch_b8
    v16bf a = load_frag(pa + k0, Kd);
#pragma unroll
    for (int t = 0; t < 4; ++t) {
      v16bf b = load_frag(Bt + (size_t)(n0 + 16 * t) * Kd + k0, Kd);
      acc[t] = wmma_bf16(a, b, acc[t]);
    }
  }

  if constexpr (MODE == 2) {
    float* o = (float*)out;
#pragma unroll
    for (int t = 0; t < 4; ++t)
#pragma unroll
      for (int v = 0; v < 8; ++v)
        o[(size_t)(row0 + v + 8 * g) * N + (n0 + 16 * t + c)] = acc[t][v];
  } else {
    __bf16* o = (__bf16*)out;
#pragma unroll
    for (int t = 0; t < 4; ++t)
#pragma unroll
      for (int v = 0; v < 8; ++v) {
        const int m = row0 + v + 8 * g;
        const int n = n0 + 16 * t + c;
        const int bb = m >> 11, tt = m & (T_ - 1);
        const int h = n >> 6, dk = n & (DK_ - 1);
        size_t idx = (MODE == 0)
                         ? ((size_t)((bb * H_ + h) * T_ + tt) * DK_ + dk)
                         : ((size_t)((bb * H_ + h) * DK_ + dk) * T_ + tt);
        o[idx] = (__bf16)acc[t][v];
      }
  }
}

// ---------------------------------------------------------------------------
// Flash attention: grid (T/64, B*H), block 128 (4 waves, 16 q-rows each).
// Q,K: [(b,h),T,DK] bf16;  Vt: [(b,h),DK,T] bf16;  ctx out: [b*T, D] bf16.
// 64-key tiles; K/V staged block-cooperatively in LDS via async copies.
// Causal masking needed only on the final (diagonal) tile of each block.
// ---------------------------------------------------------------------------
__global__ __launch_bounds__(128) void attn_kernel(
    const __bf16* __restrict__ Q, const __bf16* __restrict__ Km,
    const __bf16* __restrict__ Vt, const unsigned char* __restrict__ pmask,
    __bf16* __restrict__ ctx) {
  __shared__ __align__(16) __bf16 Ks[64 * 64];       // K rows kv..kv+63
  __shared__ __align__(16) __bf16 Vs[64 * 64];       // Vt rows d=0..63, cols kv..kv+63
  __shared__ __align__(16) __bf16 Pl[4][16 * 64];    // per-wave P staging

  const int tid  = threadIdx.x;
  const int w    = tid >> 5;
  const int lane = tid & 31;
  const int c    = lane & 15;
  const int g    = lane >> 4;
  const int bh   = blockIdx.y;
  const int b    = bh >> 4;       // H_ == 16
  const int h    = bh & 15;
  const int Q0   = blockIdx.x * 64;
  const int q0   = Q0 + w * 16;

  const __bf16* Qp = Q  + (size_t)bh * T_ * DK_;
  const __bf16* Kp = Km + (size_t)bh * T_ * DK_;
  const __bf16* Vp = Vt + (size_t)bh * DK_ * T_;
  const unsigned char* pmb = pmask + b * T_;

  const v16bf aQ0 = load_frag(Qp + q0 * DK_, DK_);
  const v16bf aQ1 = load_frag(Qp + q0 * DK_ + 32, DK_);

  v8f o[4];
#pragma unroll
  for (int t = 0; t < 4; ++t) o[t] = zero8();
  float mr[8], lr[8];
#pragma unroll
  for (int v = 0; v < 8; ++v) { mr[v] = -1e30f; lr[v] = 0.0f; }

  const float scale = 0.125f;  // 1/sqrt(DK)

  for (int kv = 0; kv <= Q0; kv += 64) {
    __syncthreads();  // previous tile fully consumed
    // ---- stage K tile (contiguous 8KB) and V tile (64 strided rows) ----
    {
      const __bf16* gk = Kp + kv * DK_;
#pragma unroll
      for (int j = 0; j < 4; ++j) {
        const int off = (tid + 128 * j) * 8;     // 16B chunks
        async_copy16(gk + off, &Ks[off]);
      }
#pragma unroll
      for (int j = 0; j < 4; ++j) {
        const int chunk = tid + 128 * j;         // 0..511
        const int drow  = chunk >> 3;
        const int coff  = (chunk & 7) * 8;
        async_copy16(Vp + drow * T_ + kv + coff, &Vs[drow * 64 + coff]);
      }
    }
    async_wait_all();
    __syncthreads();  // staged data visible to all waves

    // ---- S = Q . K^T : 4 col-tiles x 2 k-steps = 8 WMMAs ----
    v8f s[4];
#pragma unroll
    for (int j = 0; j < 4; ++j) {
      s[j] = zero8();
      s[j] = wmma_bf16(aQ0, load_frag(Ks + (16 * j) * 64, 64), s[j]);
      s[j] = wmma_bf16(aQ1, load_frag(Ks + (16 * j) * 64 + 32, 64), s[j]);
    }

    const bool edge = (kv == Q0);                // only diagonal tile masks
    bool pmv[4];
#pragma unroll
    for (int j = 0; j < 4; ++j) pmv[j] = pmb[kv + 16 * j + c] != 0;

    float alpha[8];
#pragma unroll
    for (int v = 0; v < 8; ++v) {
      const int row = q0 + v + 8 * g;
      float e[4];
#pragma unroll
      for (int j = 0; j < 4; ++j) {
        e[j] = s[j][v] * scale;
        if (pmv[j]) e[j] = -1e30f;
      }
      if (edge) {
#pragma unroll
        for (int j = 0; j < 4; ++j)
          if (kv + 16 * j + c > row) e[j] = -1e30f;
      }
      float rm = fmaxf(fmaxf(e[0], e[1]), fmaxf(e[2], e[3]));
      rm = fmaxf(rm, __shfl_xor(rm, 1, 16));
      rm = fmaxf(rm, __shfl_xor(rm, 2, 16));
      rm = fmaxf(rm, __shfl_xor(rm, 4, 16));
      rm = fmaxf(rm, __shfl_xor(rm, 8, 16));
      const float mn = fmaxf(mr[v], rm);
      alpha[v] = __expf(mr[v] - mn);
      float rs = 0.0f;
#pragma unroll
      for (int j = 0; j < 4; ++j) {
        const float p = __expf(e[j] - mn);
        rs += p;
        Pl[w][(v + 8 * g) * 64 + 16 * j + c] = (__bf16)p;
      }
      rs += __shfl_xor(rs, 1, 16);
      rs += __shfl_xor(rs, 2, 16);
      rs += __shfl_xor(rs, 4, 16);
      rs += __shfl_xor(rs, 8, 16);
      lr[v] = lr[v] * alpha[v] + rs;
      mr[v] = mn;
    }
#pragma unroll
    for (int t = 0; t < 4; ++t)
#pragma unroll
      for (int v = 0; v < 8; ++v) o[t][v] *= alpha[v];

    // ---- O += P (16x64) . V (64x64) : 2 A-frags x 4 d-blocks = 8 WMMAs ----
    const v16bf aP0 = load_frag(&Pl[w][0], 64);
    const v16bf aP1 = load_frag(&Pl[w][32], 64);
#pragma unroll
    for (int t = 0; t < 4; ++t) {
      o[t] = wmma_bf16(aP0, load_frag(Vs + (16 * t) * 64, 64), o[t]);
      o[t] = wmma_bf16(aP1, load_frag(Vs + (16 * t) * 64 + 32, 64), o[t]);
    }
  }

  // normalize and write ctx (bf16, [b*T, D] row-major)
#pragma unroll
  for (int v = 0; v < 8; ++v) {
    const float inv = 1.0f / lr[v];
    const int trow = q0 + v + 8 * g;
#pragma unroll
    for (int t = 0; t < 4; ++t)
      ctx[(size_t)(b * T_ + trow) * D_ + h * DK_ + t * 16 + c] =
          (__bf16)(o[t][v] * inv);
  }
}

// ---------------------------------------------------------------------------
// Host-side launch
// ---------------------------------------------------------------------------
extern "C" void kernel_launch(void* const* d_in, const int* in_sizes, int n_in,
                              void* d_out, int out_size, void* d_ws,
                              size_t ws_size, hipStream_t stream) {
  (void)in_sizes; (void)n_in; (void)out_size; (void)ws_size;

  const float* X1 = (const float*)d_in[0];
  const float* X2 = (const float*)d_in[1];
  const unsigned char* pmask = (const unsigned char*)d_in[2];
  const float* WQ = (const float*)d_in[3];
  const float* WK = (const float*)d_in[4];
  const float* WV = (const float*)d_in[5];
  const float* WO = (const float*)d_in[6];
  float* out = (float*)d_out;

  const int M = B_ * T_;             // 4096
  const int NX = M * D_;             // elements in one X
  const size_t MB = 1u << 20;

  char* ws = (char*)d_ws;
  __bf16* X1bf = (__bf16*)(ws + 0 * MB);
  __bf16* X2bf = (__bf16*)(ws + 8 * MB);
  __bf16* WQt  = (__bf16*)(ws + 16 * MB);
  __bf16* WKt  = (__bf16*)(ws + 18 * MB);
  __bf16* WVt  = (__bf16*)(ws + 20 * MB);
  __bf16* WOt  = (__bf16*)(ws + 22 * MB);
  __bf16* Qbf  = (__bf16*)(ws + 24 * MB);
  __bf16* Kbf  = (__bf16*)(ws + 32 * MB);
  __bf16* Vtb  = (__bf16*)(ws + 40 * MB);
  __bf16* Ctx  = (__bf16*)(ws + 48 * MB);

  // 1) convert activations to bf16
  cvt_bf16_kernel<<<NX / 1024, 256, 0, stream>>>(X1, X1bf, NX);
  cvt_bf16_kernel<<<NX / 1024, 256, 0, stream>>>(X2, X2bf, NX);

  // 2) convert weights to transposed bf16
  dim3 tg(D_ / 32, D_ / 32), tb(32, 8);
  wtrans_kernel<<<tg, tb, 0, stream>>>(WQ, WQt);
  wtrans_kernel<<<tg, tb, 0, stream>>>(WK, WKt);
  wtrans_kernel<<<tg, tb, 0, stream>>>(WV, WVt);
  wtrans_kernel<<<tg, tb, 0, stream>>>(WO, WOt);

  // 3) projections (WMMA):  Q,K head-split row-major;  V transposed
  dim3 gg(D_ / 128, M / 64);
  gemm_bf16_kernel<0><<<gg, 256, 0, stream>>>(X1bf, WQt, (void*)Qbf, M, D_, D_);
  gemm_bf16_kernel<0><<<gg, 256, 0, stream>>>(X2bf, WKt, (void*)Kbf, M, D_, D_);
  gemm_bf16_kernel<1><<<gg, 256, 0, stream>>>(X2bf, WVt, (void*)Vtb, M, D_, D_);

  // 4) flash attention
  attn_kernel<<<dim3(T_ / 64, B_ * H_), 128, 0, stream>>>(Qbf, Kbf, Vtb, pmask,
                                                          Ctx);

  // 5) output projection -> f32 d_out
  gemm_bf16_kernel<2><<<gg, 256, 0, stream>>>(Ctx, WOt, (void*)out, M, D_, D_);
}
